// SinkhornRouter_49916109914335
// MI455X (gfx1250) — matrix operand order repaired and checked by reading, not compiled, not run on hardware
//
#include <hip/hip_runtime.h>
#include <hip/hip_bf16.h>

typedef __attribute__((ext_vector_type(16))) _Float16 v16h;
typedef __attribute__((ext_vector_type(8)))  float    v8f;

#define B_ 4
#define N_ 4096
#define D_ 1024
#define E_ 8
#define M_ 512           // tokens per expert (top-k)

__device__ __forceinline__ unsigned short f2h(float f) {
    union { _Float16 h; unsigned short u; } c;
    c.h = (_Float16)f;
    return c.u;
}

union FragU { uint4 u[2]; v16h v; };

// CDNA5 async copy: global memory -> LDS, no VGPR round trip, ASYNCcnt-tracked.
__device__ __forceinline__ void async_b128(unsigned lds_byte_off, const void* gaddr) {
    asm volatile("global_load_async_to_lds_b128 %0, %1, off"
                 :: "v"(lds_byte_off),
                    "v"((unsigned long long)(uintptr_t)gaddr)
                 : "memory");
}

// ---------------------------------------------------------------- kernel 0a
// x fp32 -> f16 (row-major), one-time; makes GEMM A-staging a pure async copy.
__global__ __launch_bounds__(256)
void convert_x_kernel(const float* __restrict__ x, unsigned short* __restrict__ xh) {
    size_t n4 = (size_t)B_ * N_ * D_ / 4;
    size_t stride = (size_t)gridDim.x * blockDim.x;
    for (size_t i = (size_t)blockIdx.x * blockDim.x + threadIdx.x; i < n4; i += stride) {
        float4 f = ((const float4*)x)[i];
        unsigned u0 = (unsigned)f2h(f.x) | ((unsigned)f2h(f.y) << 16);
        unsigned u1 = (unsigned)f2h(f.z) | ((unsigned)f2h(f.w) << 16);
        ((uint2*)xh)[i] = make_uint2(u0, u1);
    }
}

// ---------------------------------------------------------------- kernel 0b
// experts[e][k][n] fp32 -> wT[e][n][k] f16 (col-major for B fragments).
__global__ __launch_bounds__(256)
void convert_wT_kernel(const float* __restrict__ w, unsigned short* __restrict__ wT) {
    __shared__ unsigned short tile[32][33];
    const int e  = blockIdx.z;
    const int kb = blockIdx.x * 32;
    const int nb = blockIdx.y * 32;
    const int tx = threadIdx.x & 31;
    const int ty = threadIdx.x >> 5;     // 0..7
    const float* we = w + (size_t)e * D_ * D_;
    #pragma unroll
    for (int r = 0; r < 32; r += 8)
        tile[ty + r][tx] = f2h(we[(size_t)(kb + ty + r) * D_ + nb + tx]);
    __syncthreads();
    unsigned short* wTe = wT + (size_t)e * D_ * D_;
    #pragma unroll
    for (int r = 0; r < 32; r += 8)
        wTe[(size_t)(nb + ty + r) * D_ + kb + tx] = tile[tx][ty + r];
}

// ---------------------------------------------------------------- kernel 1
// gate logits: (16384 x 1024) x (1024 x 8)
__global__ __launch_bounds__(256)
void gate_logits_kernel(const float* __restrict__ x,
                        const float* __restrict__ gw,
                        float* __restrict__ logits) {
    __shared__ float sgw[D_ * E_];     // 32 KB
    for (int i = threadIdx.x; i < D_ * E_; i += 256) sgw[i] = gw[i];
    __syncthreads();
    int tok = blockIdx.x * 256 + threadIdx.x;     // 0 .. 16383
    const float* xr = x + (size_t)tok * D_;
    float acc[E_];
    #pragma unroll
    for (int e = 0; e < E_; ++e) acc[e] = 0.0f;
    for (int k = 0; k < D_; k += 4) {
        float4 xv = *(const float4*)(xr + k);
        #pragma unroll
        for (int e = 0; e < E_; ++e) {
            acc[e] += xv.x * sgw[(k + 0) * E_ + e];
            acc[e] += xv.y * sgw[(k + 1) * E_ + e];
            acc[e] += xv.z * sgw[(k + 2) * E_ + e];
            acc[e] += xv.w * sgw[(k + 3) * E_ + e];
        }
    }
    #pragma unroll
    for (int e = 0; e < E_; ++e) logits[(size_t)tok * E_ + e] = acc[e];
}

// ---------------------------------------------------------------- kernel 2
// Sinkhorn: one block per batch, whole (4096 x 8) matrix in registers.
__global__ __launch_bounds__(1024)
void sinkhorn_kernel(float* __restrict__ g) {
    __shared__ float red_m[32 * E_];
    __shared__ float red_s[32 * E_];
    const int bb = blockIdx.x;
    const int tid = threadIdx.x;
    const int lane = tid & 31;
    const int wave = tid >> 5;
    float* gb = g + (size_t)bb * N_ * E_;

    float t[4][E_];
    #pragma unroll
    for (int r = 0; r < 4; ++r) {
        int row = tid + r * 1024;
        #pragma unroll
        for (int e = 0; e < E_; ++e)
            t[r][e] = logf(fmaxf(gb[(size_t)row * E_ + e], 1e-6f));
    }

    for (int it = 0; it < 8; ++it) {
        // ---- logsumexp over n (axis -2), per expert column
        float m[E_], s[E_];
        #pragma unroll
        for (int e = 0; e < E_; ++e) {
            m[e] = t[0][e]; s[e] = 1.0f;
            #pragma unroll
            for (int r = 1; r < 4; ++r) {
                float M = fmaxf(m[e], t[r][e]);
                s[e] = s[e] * expf(m[e] - M) + expf(t[r][e] - M);
                m[e] = M;
            }
            #pragma unroll
            for (int off = 16; off > 0; off >>= 1) {
                float m2 = __shfl_xor(m[e], off, 32);
                float s2 = __shfl_xor(s[e], off, 32);
                float M = fmaxf(m[e], m2);
                s[e] = s[e] * expf(m[e] - M) + s2 * expf(m2 - M);
                m[e] = M;
            }
        }
        if (lane == 0) {
            #pragma unroll
            for (int e = 0; e < E_; ++e) { red_m[wave * E_ + e] = m[e]; red_s[wave * E_ + e] = s[e]; }
        }
        __syncthreads();
        float lse[E_];
        #pragma unroll
        for (int e = 0; e < E_; ++e) {
            float M = red_m[e], S = red_s[e];
            for (int w = 1; w < 32; ++w) {
                float m2 = red_m[w * E_ + e], s2 = red_s[w * E_ + e];
                float MM = fmaxf(M, m2);
                S = S * expf(M - MM) + s2 * expf(m2 - MM);
                M = MM;
            }
            lse[e] = M + logf(S);
        }
        __syncthreads();
        #pragma unroll
        for (int r = 0; r < 4; ++r)
            #pragma unroll
            for (int e = 0; e < E_; ++e) t[r][e] -= lse[e];

        // ---- logsumexp over e (axis -1), thread-local
        #pragma unroll
        for (int r = 0; r < 4; ++r) {
            float M = t[r][0];
            #pragma unroll
            for (int e = 1; e < E_; ++e) M = fmaxf(M, t[r][e]);
            float S = 0.0f;
            #pragma unroll
            for (int e = 0; e < E_; ++e) S += expf(t[r][e] - M);
            float l = M + logf(S);
            #pragma unroll
            for (int e = 0; e < E_; ++e) t[r][e] -= l;
        }
    }

    #pragma unroll
    for (int r = 0; r < 4; ++r) {
        int row = tid + r * 1024;
        #pragma unroll
        for (int e = 0; e < E_; ++e)
            gb[(size_t)row * E_ + e] = expf(t[r][e]);
    }
}

// ---------------------------------------------------------------- kernel 3
__global__ void init_kernel(float* __restrict__ out, int* __restrict__ owner, size_t n4) {
    size_t gid = (size_t)blockIdx.x * blockDim.x + threadIdx.x;
    size_t stride = (size_t)gridDim.x * blockDim.x;
    float4 z = make_float4(0.f, 0.f, 0.f, 0.f);
    for (size_t i = gid; i < n4; i += stride) ((float4*)out)[i] = z;
    if (gid < (size_t)B_ * N_) owner[gid] = -1;
}

// ---------------------------------------------------------------- kernel 4
// Bitonic top-512 of 4096 per (batch, expert). Highest expert wins scatter.
__global__ __launch_bounds__(1024)
void topk_kernel(const float* __restrict__ gates, int* __restrict__ ridx,
                 float* __restrict__ hard, int* __restrict__ owner) {
    __shared__ float sv[N_];
    __shared__ int   si[N_];
    const int be = blockIdx.x;
    const int bb = be >> 3;
    const int e  = be & 7;
    const float* gb = gates + (size_t)bb * N_ * E_;
    const int tid = threadIdx.x;
    for (int i = tid; i < N_; i += 1024) { sv[i] = gb[(size_t)i * E_ + e]; si[i] = i; }
    __syncthreads();
    for (int kk = 2; kk <= N_; kk <<= 1) {
        for (int j = kk >> 1; j > 0; j >>= 1) {
            for (int i = tid; i < N_; i += 1024) {
                int p = i ^ j;
                if (p > i) {
                    bool up = ((i & kk) == 0);
                    float v1 = sv[i], v2 = sv[p];
                    int i1 = si[i], i2 = si[p];
                    bool i_after_p = (v1 < v2) || (v1 == v2 && i1 > i2);
                    if (i_after_p == up) {
                        sv[i] = v2; sv[p] = v1; si[i] = i2; si[p] = i1;
                    }
                }
            }
            __syncthreads();
        }
    }
    for (int mth = tid; mth < M_; mth += 1024) {
        int idx = si[mth];
        ridx[be * M_ + mth] = idx;
        hard[be * M_ + mth] = sv[mth] > 0.5f ? 1.0f : 0.0f;
        atomicMax(&owner[bb * N_ + idx], e);
    }
}

// ---------------------------------------------------------------- kernel 5
// Expert GEMM: per (b,e): Y(512x1024) = gather(xh)(512x1024) @ Wt(1024x1024)
// 128x128 tile/block, 8 waves of 32x64, WMMA f32_16x16x32_f16.
// Double-buffered LDS fed by global_load_async_to_lds_b128 (ASYNCcnt).
#define TM 128
#define TN 128
#define TK 32

__global__ __launch_bounds__(256)
void expert_gemm_kernel(const unsigned short* __restrict__ xh,
                        const unsigned short* __restrict__ wT,
                        const int* __restrict__ ridx, const float* __restrict__ hard,
                        const int* __restrict__ owner, float* __restrict__ out) {
    __shared__ __align__(16) unsigned short lds_a[2][TM * TK];  // row-major [row][k]
    __shared__ __align__(16) unsigned short lds_b[2][TN * TK];  // col-major [col][k]
    __shared__ int   lds_tok[TM];
    __shared__ float lds_hard[TM];

    const int tid  = threadIdx.x;
    const int lane = tid & 31;
    const int wave = tid >> 5;     // 0..7
    const int wm   = wave & 3;     // 4 M-blocks of 32 rows
    const int wn   = wave >> 2;    // 2 N-blocks of 64 cols

    const int n0 = blockIdx.x * TN;      // N tile base
    const int m0 = blockIdx.y * TM;      // M tile base (of 512 routed)
    const int be = blockIdx.z;           // 0..31
    const int bb = be >> 3;
    const int e  = be & 7;

    if (tid < TM) {
        lds_tok[tid]  = ridx[be * M_ + m0 + tid];
        lds_hard[tid] = hard[be * M_ + m0 + tid];
    }
    __syncthreads();

    // Per-thread async-copy descriptors: 512 chunks of 16B per matrix tile,
    // chunk c -> row/col = c>>2, q = c&3 (16B at k-offset q*8 halves).
    const int c0 = tid, c1 = tid + 256;
    const int r0 = c0 >> 2, q0 = c0 & 3;
    const int r1 = c1 >> 2, q1 = c1 & 3;

    const unsigned short* xb = xh + (size_t)bb * N_ * D_;
    const unsigned short* wb = wT + (size_t)e * D_ * D_;

    const unsigned short* ag0 = xb + (size_t)lds_tok[r0] * D_ + q0 * 8;
    const unsigned short* ag1 = xb + (size_t)lds_tok[r1] * D_ + q1 * 8;
    const unsigned short* bg0 = wb + (size_t)(n0 + r0) * D_ + q0 * 8;
    const unsigned short* bg1 = wb + (size_t)(n0 + r1) * D_ + q1 * 8;

    unsigned al0[2], al1[2], bl0[2], bl1[2];
    #pragma unroll
    for (int b = 0; b < 2; ++b) {
        al0[b] = (unsigned)(uintptr_t)&lds_a[b][r0 * TK + q0 * 8];
        al1[b] = (unsigned)(uintptr_t)&lds_a[b][r1 * TK + q1 * 8];
        bl0[b] = (unsigned)(uintptr_t)&lds_b[b][r0 * TK + q0 * 8];
        bl1[b] = (unsigned)(uintptr_t)&lds_b[b][r1 * TK + q1 * 8];
    }

    auto issue_tile = [&](int ks, int buf) {
        async_b128(al0[buf], ag0 + ks);
        async_b128(al1[buf], ag1 + ks);
        async_b128(bl0[buf], bg0 + ks);
        async_b128(bl1[buf], bg1 + ks);
    };

    v8f acc[2][4];
    #pragma unroll
    for (int mt = 0; mt < 2; ++mt)
        #pragma unroll
        for (int nt = 0; nt < 4; ++nt)
            acc[mt][nt] = (v8f){0.f, 0.f, 0.f, 0.f, 0.f, 0.f, 0.f, 0.f};

    issue_tile(0, 0);                       // prologue: tile 0 in flight

    for (int ks = 0; ks < D_; ks += TK) {
        const int cur = (ks >> 5) & 1;
        if (ks + TK < D_) {
            issue_tile(ks + TK, cur ^ 1);   // prefetch next tile (4 ops/wave)
            asm volatile("s_wait_asynccnt 0x4" ::: "memory");  // current tile done
        } else {
            asm volatile("s_wait_asynccnt 0x0" ::: "memory");
        }
        __syncthreads();                    // publish current tile to all waves

        const int kh  = lane >> 4;          // lane half
        const int kb1 = kh * 8;             // A: elems 0..7  -> K kb1..kb1+7
        const int kb2 = 16 + kh * 8;        // A: elems 8..15 -> K kb2..kb2+7
        v16h afrag[2], bfrag[4];
        #pragma unroll
        for (int mt = 0; mt < 2; ++mt) {
            int row = wm * 32 + mt * 16 + (lane & 15);
            FragU f;
            f.u[0] = *(const uint4*)&lds_a[cur][row * TK + kb1];
            f.u[1] = *(const uint4*)&lds_a[cur][row * TK + kb2];
            afrag[mt] = f.v;
        }
        #pragma unroll
        for (int nt = 0; nt < 4; ++nt) {
            int col = wn * 64 + nt * 16 + (lane & 15);
            const unsigned short* p = &lds_b[cur][col * TK + kh * 8]; // B: same K order as A
            FragU f;
            f.u[0] = *(const uint4*)(p);
            f.u[1] = *(const uint4*)(p + 16);
            bfrag[nt] = f.v;
        }
        #pragma unroll
        for (int mt = 0; mt < 2; ++mt)
            #pragma unroll
            for (int nt = 0; nt < 4; ++nt)
                acc[mt][nt] = __builtin_amdgcn_wmma_f32_16x16x32_f16(
                    false, afrag[mt], false, bfrag[nt],
                    (short)0, acc[mt][nt], false, false);

        __syncthreads();   // all reads of `cur` done before it is overwritten
    }

    // ---- epilogue: scale by hard gate, scatter rows this expert owns
    #pragma unroll
    for (int mt = 0; mt < 2; ++mt) {
        #pragma unroll
        for (int i = 0; i < 8; ++i) {
            int rloc = wm * 32 + mt * 16 + i + (lane >> 4) * 8;
            int tok  = lds_tok[rloc];
            float h  = lds_hard[rloc];
            if (owner[bb * N_ + tok] == e) {
                size_t base = ((size_t)bb * N_ + tok) * D_;
                #pragma unroll
                for (int nt = 0; nt < 4; ++nt)
                    out[base + n0 + wn * 64 + nt * 16 + (lane & 15)] = acc[mt][nt][i] * h;
            }
        }
    }
}

// ---------------------------------------------------------------- launch
extern "C" void kernel_launch(void* const* d_in, const int* in_sizes, int n_in,
                              void* d_out, int out_size, void* d_ws, size_t ws_size,
                              hipStream_t stream) {
    (void)in_sizes; (void)n_in; (void)out_size; (void)ws_size;
    const float* x       = (const float*)d_in[0];   // (4,4096,1024)
    const float* gw      = (const float*)d_in[1];   // (1,1024,8)
    const float* experts = (const float*)d_in[2];   // (8,1024,1024)
    float* out = (float*)d_out;                     // (4,4096,1024)

    char* ws = (char*)d_ws;
    unsigned short* xh    = (unsigned short*)ws;                  // 32 MB f16 x
    unsigned short* wT    = (unsigned short*)(ws + 33554432);     // 16 MB f16 expertsT
    float* gates = (float*)(ws + 50331648);                       // 512 KB
    int*   ridx  = (int*)  (ws + 50855936);                       //  64 KB
    float* hard  = (float*)(ws + 50921472);                       //  64 KB
    int*   owner = (int*)  (ws + 50987008);                       //  64 KB

    convert_x_kernel<<<2048, 256, 0, stream>>>(x, xh);
    convert_wT_kernel<<<dim3(32, 32, 8), 256, 0, stream>>>(experts, wT);
    gate_logits_kernel<<<(B_ * N_) / 256, 256, 0, stream>>>(x, gw, gates);
    sinkhorn_kernel<<<B_, 1024, 0, stream>>>(gates);
    init_kernel<<<2048, 256, 0, stream>>>(out, owner, (size_t)B_ * N_ * D_ / 4);
    topk_kernel<<<B_ * E_, 1024, 0, stream>>>(gates, ridx, hard, owner);
    expert_gemm_kernel<<<dim3(D_ / TN, M_ / TM, B_ * E_), 256, 0, stream>>>(
        xh, wT, ridx, hard, owner, out);
}
